// HSPAM_66133906424100
// MI455X (gfx1250) — matrix operand order repaired and checked by reading, not compiled, not run on hardware
//
#include <hip/hip_runtime.h>
#include <hip/hip_bf16.h>

// ---------------------------------------------------------------------------
// HSPAM for MI455X (gfx1250, wave32, WMMA f32_16x16x32_f16)
// N=8, C=128, H=W=48 (HW=2304), CE=16, TOPK=128, NRES=4
// Conv3x3 weights repacked position-major (k' = r*128 + ci) so each WMMA
// K-step has a uniform kernel tap -> no integer division in the hot loop.
// ---------------------------------------------------------------------------

typedef __attribute__((ext_vector_type(16))) _Float16 v16h;
typedef __attribute__((ext_vector_type(8)))  _Float16 v8h;
typedef __attribute__((ext_vector_type(8)))  float    v8f;
typedef __attribute__((ext_vector_type(4)))  float    v4f;

#define NB    8
#define CC    128
#define HH    48
#define WW    48
#define HW    2304          // H*W
#define NHW   18432         // N*H*W
#define KK    1152          // C*9
#define CE    16
#define TOPKK 128
#define NRES  4
#define SRS   2308          // padded score row stride (f32): 2308 % 64 == 4

enum { MODE_RELU = 0, MODE_RES = 1, MODE_FINAL = 2 };

// ------------------------------- cast helper -------------------------------
__global__ void cast_f32_to_f16(const float* __restrict__ src,
                                _Float16* __restrict__ dst, int n) {
  int i = blockIdx.x * 256 + threadIdx.x;
  if (i < n) dst[i] = (_Float16)src[i];
}

// 3x3 weight repack: dst[co][r*128+ci] = src[co][ci][r]  (f32 -> f16)
__global__ void repack_w3x3(const float* __restrict__ src,
                            _Float16* __restrict__ dst, int total) {
  int i = blockIdx.x * 256 + threadIdx.x;
  if (i >= total) return;
  int co = i / KK;
  int k  = i - co * KK;
  int r  = k >> 7;          // kernel tap 0..8
  int ci = k & 127;
  dst[i] = (_Float16)src[((long)co * CC + ci) * 9 + r];
}

// ------------------------- conv3x3 implicit-GEMM ---------------------------
// grid = (NHW/16, C/64), block = 128 (4 waves). wave w owns 16 output chans.
__global__ __launch_bounds__(128) void conv3x3_wmma(
    const _Float16* __restrict__ in,     // (N,C,H,W) f16
    const _Float16* __restrict__ wpack,  // (Co, 9*128) position-major f16
    const float* __restrict__ bias,      // (Co)
    int mode,
    const _Float16* __restrict__ res,    // residual (MODE_RES)
    const float* __restrict__ xin,       // (MODE_FINAL)
    _Float16* __restrict__ out16,
    float* __restrict__ out32) {
  __shared__ _Float16 As[16][40];        // [pixel][k_local], padded row

  const int t    = threadIdx.x;
  const int lane = t & 31, wave = t >> 5;
  const int hid  = lane >> 4, l16 = lane & 15;

  const int gp  = blockIdx.x * 16;       // global pixel base (row-aligned)
  const int n   = gp / HW;
  const int rem = gp - n * HW;           // pixel-in-image base
  const int y   = rem / WW, x0 = rem - y * WW;
  const int co  = blockIdx.y * 64 + wave * 16 + l16;

  const _Float16* wrow = wpack + (long)co * KK;
  v8f acc = {};

  for (int r = 0; r < 9; ++r) {          // kernel tap (uniform per step)
    const int dy = r / 3 - 1, dx = r % 3 - 1;   // tiny, once per 4 K-steps
    const int yy = y + dy;
    const bool yok = (yy >= 0) & (yy < HH);
    const _Float16* inb = in + ((long)n * CC) * HW + yy * WW;

    for (int cb = 0; cb < 128; cb += 32) {
      const int kb = r * 128 + cb;
      // stage As[p][kl] = in[n, cb+kl, yy, x0+p+dx]
#pragma unroll
      for (int i = 0; i < 4; ++i) {
        int idx = i * 128 + t;
        int kl = idx >> 4, p = idx & 15;
        int ci = cb + kl;
        int xx = x0 + p + dx;
        float v = 0.f;
        if (yok && xx >= 0 && xx < WW)
          v = (float)inb[(long)ci * HW + xx];
        As[p][kl] = (_Float16)v;
      }
      __syncthreads();

      v8h lo = *(const v8h*)&As[l16][hid * 16];
      v8h hi = *(const v8h*)&As[l16][hid * 16 + 8];
      v16h a = __builtin_shufflevector(lo, hi, 0, 1, 2, 3, 4, 5, 6, 7,
                                       8, 9, 10, 11, 12, 13, 14, 15);
      __builtin_prefetch(wrow + kb + 64, 0, 0);   // global_prefetch_b8
      v16h b = *(const v16h*)(wrow + kb + hid * 16);
      acc = __builtin_amdgcn_wmma_f32_16x16x32_f16(false, a, false, b,
                                                   (short)0, acc, false, false);
      __syncthreads();
    }
  }

  const float bco = bias[co];
#pragma unroll
  for (int r = 0; r < 8; ++r) {
    int m = r + 8 * hid;                 // D row -> pixel offset in tile
    long idx = (long)(n * CC + co) * HW + (rem + m);
    float v = acc[r] + bco;
    if (mode == MODE_RELU) {
      out16[idx] = (_Float16)fmaxf(v, 0.f);
    } else if (mode == MODE_RES) {
      v += (float)res[idx];
      out16[idx] = (_Float16)v;
    } else {
      out32[idx] = v * 0.1f + xin[idx];
    }
  }
}

// ----------------------- q/k/v 1x1 convs + PReLU ---------------------------
__device__ __forceinline__ v8f gemm16_k128(const _Float16 (*Sa)[136],
                                           const _Float16* wrow,
                                           int hid, int l16) {
  v8f acc = {};
#pragma unroll
  for (int ks = 0; ks < 128; ks += 32) {
    v8h lo = *(const v8h*)&Sa[l16][ks + hid * 16];
    v8h hi = *(const v8h*)&Sa[l16][ks + hid * 16 + 8];
    v16h a = __builtin_shufflevector(lo, hi, 0, 1, 2, 3, 4, 5, 6, 7,
                                     8, 9, 10, 11, 12, 13, 14, 15);
    v16h b = *(const v16h*)(wrow + ks + hid * 16);
    acc = __builtin_amdgcn_wmma_f32_16x16x32_f16(false, a, false, b,
                                                 (short)0, acc, false, false);
  }
  return acc;
}

// grid = NHW/16, block = 128. wave0->q, wave1->k, waves2,3->v (4 tiles each).
__global__ __launch_bounds__(128) void qkv_wmma(
    const _Float16* __restrict__ act,
    const _Float16* __restrict__ wq, const _Float16* __restrict__ wk,
    const _Float16* __restrict__ wv,
    const float* __restrict__ bq, const float* __restrict__ bk,
    const float* __restrict__ bv,
    const float* __restrict__ pq, const float* __restrict__ pk,
    const float* __restrict__ pv,
    _Float16* __restrict__ qbuf,   // (N,HW,32) ch-padded, upper 16 pre-zeroed
    _Float16* __restrict__ kbuf,   // (N,HW,32)
    _Float16* __restrict__ vbuf) { // (N,C,HW)
  __shared__ _Float16 Sa[16][136];       // [pixel][ci], padded row

  const int t    = threadIdx.x;
  const int lane = t & 31, wave = t >> 5;
  const int hid  = lane >> 4, l16 = lane & 15;
  const int gp   = blockIdx.x * 16;
  const int n    = gp / HW;
  const int rem  = gp - n * HW;

#pragma unroll
  for (int i = 0; i < 16; ++i) {         // stage act tile: Sa[pixel][ci]
    int idx = i * 128 + t;
    int ci = idx >> 4, p = idx & 15;
    Sa[p][ci] = act[((long)(n * CC + ci)) * HW + rem + p];
  }
  __syncthreads();

  if (wave < 2) {
    const _Float16* w = (wave == 0) ? wq : wk;
    const float* bb   = (wave == 0) ? bq : bk;
    const float slope = (wave == 0) ? pq[0] : pk[0];
    _Float16* obuf    = (wave == 0) ? qbuf : kbuf;
    v8f acc = gemm16_k128(Sa, w + (long)l16 * 128, hid, l16);
    float bco = bb[l16];
#pragma unroll
    for (int r = 0; r < 8; ++r) {
      float v = acc[r] + bco;
      v = (v >= 0.f) ? v : v * slope;
      int pix = rem + r + 8 * hid;
      obuf[((long)n * HW + pix) * 32 + l16] = (_Float16)v;
    }
  } else {
    const float slope = pv[0];
    for (int ct = 0; ct < 4; ++ct) {
      int co = (wave - 2) * 64 + ct * 16 + l16;
      v8f acc = gemm16_k128(Sa, wv + (long)co * 128, hid, l16);
      float bco = bv[co];
#pragma unroll
      for (int r = 0; r < 8; ++r) {
        float v = acc[r] + bco;
        v = (v >= 0.f) ? v : v * slope;
        int pix = rem + r + 8 * hid;
        vbuf[((long)(n * CC + co)) * HW + pix] = (_Float16)v;
      }
    }
  }
}

// ----------------- fused HSPA: scores -> top-k -> softmax -> AV ------------
// grid = N*(HW/16) = 1152, block = 128. Dynamic LDS: 16*2308 f32 (~144KB).
__global__ __launch_bounds__(128) void hspa_attn(
    const _Float16* __restrict__ qbuf, const _Float16* __restrict__ kbuf,
    const _Float16* __restrict__ vbuf, const _Float16* __restrict__ hact,
    _Float16* __restrict__ h2) {
  extern __shared__ float S[];           // [16][SRS] score strip (padded)
  __shared__ float Red[128];
  __shared__ float rowMax[16], rowLo[16], rowHi[16], rowThr[16], rowInv[16];

  const int t    = threadIdx.x;
  const int lane = t & 31, wave = t >> 5;
  const int hid  = lane >> 4, l16 = lane & 15;
  const int b    = blockIdx.x / (HW / 16);
  const int qt   = blockIdx.x % (HW / 16);
  const long qpix0 = (long)b * HW + qt * 16;

  // ---- stage 1: S = q . k^T (K padded 16->32 with zeros) ----
  v16h qa;
  {
    const _Float16* qp = qbuf + (qpix0 + l16) * 32;
    v8h lo = *(const v8h*)(qp + hid * 16);
    v8h hi = *(const v8h*)(qp + hid * 16 + 8);
    qa = __builtin_shufflevector(lo, hi, 0, 1, 2, 3, 4, 5, 6, 7,
                                 8, 9, 10, 11, 12, 13, 14, 15);
  }
  for (int kt = wave; kt < HW / 16; kt += 4) {
    v16h kb = *(const v16h*)(kbuf + ((long)b * HW + kt * 16 + l16) * 32 + hid * 16);
    v8f c = {};
    c = __builtin_amdgcn_wmma_f32_16x16x32_f16(false, qa, false, kb,
                                               (short)0, c, false, false);
#pragma unroll
    for (int r = 0; r < 8; ++r)
      S[(r + 8 * hid) * SRS + kt * 16 + l16] = c[r];
  }
  __syncthreads();

  // ---- stage 2: per-row max/min, binary-search top-128 threshold, sumexp --
  const int row = t >> 3, sub = t & 7;
  float lmax = -3.4e38f, lmin = 3.4e38f;
  for (int j = sub; j < HW; j += 8) {
    float s = S[row * SRS + j];
    lmax = fmaxf(lmax, s);
    lmin = fminf(lmin, s);
  }
  Red[t] = lmax; __syncthreads();
  if (sub == 0) {
    float m = Red[row * 8];
    for (int i = 1; i < 8; ++i) m = fmaxf(m, Red[row * 8 + i]);
    rowMax[row] = m; rowHi[row] = m + 1.0f;
  }
  __syncthreads();
  Red[t] = lmin; __syncthreads();
  if (sub == 0) {
    float m = Red[row * 8];
    for (int i = 1; i < 8; ++i) m = fminf(m, Red[row * 8 + i]);
    rowLo[row] = m - 1.0f;
  }
  __syncthreads();

  for (int it = 0; it < 24; ++it) {      // invariant: cnt(>=lo)>=128>cnt(>=hi)
    float mid = 0.5f * (rowLo[row] + rowHi[row]);
    int cnt = 0;
    for (int j = sub; j < HW; j += 8) cnt += (S[row * SRS + j] >= mid) ? 1 : 0;
    Red[t] = (float)cnt; __syncthreads();
    if (sub == 0) {
      int c = 0;
      for (int i = 0; i < 8; ++i) c += (int)Red[row * 8 + i];
      if (c >= TOPKK) rowLo[row] = mid; else rowHi[row] = mid;
    }
    __syncthreads();
  }
  {
    float thr = rowLo[row], mx = rowMax[row], lsum = 0.f;
    for (int j = sub; j < HW; j += 8) {
      float s = S[row * SRS + j];
      if (s >= thr) lsum += __expf(s - mx);
    }
    Red[t] = lsum; __syncthreads();
    if (sub == 0) {
      float sm = 0.f;
      for (int i = 0; i < 8; ++i) sm += Red[row * 8 + i];
      rowThr[row] = thr;
      rowInv[row] = 1.0f / sm;
    }
    __syncthreads();
  }

  // ---- stage 3: out(16q x 128ch) = P x V, fused +0.1*h epilogue ----
  const float thrR = rowThr[l16], mxR = rowMax[l16], invR = rowInv[l16];
  const int ct0 = wave * 2;              // two 16-channel tiles per wave
  v8f acc0 = {}, acc1 = {};
  const _Float16* vbase0 = vbuf + ((long)(b * CC + ct0 * 16 + l16)) * HW;
  const _Float16* vbase1 = vbase0 + (long)16 * HW;
  const float* srow = S + l16 * SRS;

  for (int j0 = 0; j0 < HW; j0 += 32) {
    // 16 K-contiguous scores per half -> aligned 16B LDS vector loads
    v4f s0 = *(const v4f*)(srow + j0 + hid * 16);
    v4f s1 = *(const v4f*)(srow + j0 + hid * 16 + 4);
    v4f s2 = *(const v4f*)(srow + j0 + hid * 16 + 8);
    v4f s3 = *(const v4f*)(srow + j0 + hid * 16 + 12);
    v16h p;
#pragma unroll
    for (int e = 0; e < 4; ++e) {
      float a0 = s0[e], a1 = s1[e], a2 = s2[e], a3 = s3[e];
      p[e]      = (_Float16)((a0 >= thrR) ? __expf(a0 - mxR) * invR : 0.f);
      p[e + 4]  = (_Float16)((a1 >= thrR) ? __expf(a1 - mxR) * invR : 0.f);
      p[e + 8]  = (_Float16)((a2 >= thrR) ? __expf(a2 - mxR) * invR : 0.f);
      p[e + 12] = (_Float16)((a3 >= thrR) ? __expf(a3 - mxR) * invR : 0.f);
    }
    v16h b0 = *(const v16h*)(vbase0 + j0 + hid * 16);
    v16h b1 = *(const v16h*)(vbase1 + j0 + hid * 16);
    acc0 = __builtin_amdgcn_wmma_f32_16x16x32_f16(false, p, false, b0,
                                                  (short)0, acc0, false, false);
    acc1 = __builtin_amdgcn_wmma_f32_16x16x32_f16(false, p, false, b1,
                                                  (short)0, acc1, false, false);
  }

#pragma unroll
  for (int r = 0; r < 8; ++r) {
    int pix = qt * 16 + r + 8 * hid;
    int ch = ct0 * 16 + l16;
    long idx0 = (long)(b * CC + ch) * HW + pix;
    long idx1 = idx0 + (long)16 * HW;
    float o0 = acc0[r] + 0.1f * (float)hact[idx0];
    float o1 = acc1[r] + 0.1f * (float)hact[idx1];
    h2[idx0] = (_Float16)o0;
    h2[idx1] = (_Float16)o1;
  }
}

// ------------------------------- launcher ----------------------------------
extern "C" void kernel_launch(void* const* d_in, const int* in_sizes, int n_in,
                              void* d_out, int out_size, void* d_ws,
                              size_t ws_size, hipStream_t stream) {
  const float* x      = (const float*)d_in[0];
  const float* leb_w1 = (const float*)d_in[1];
  const float* leb_b1 = (const float*)d_in[2];
  const float* leb_w2 = (const float*)d_in[3];
  const float* leb_b2 = (const float*)d_in[4];
  const float* wm1    = (const float*)d_in[5];
  const float* bm1    = (const float*)d_in[6];
  const float* pm1    = (const float*)d_in[7];
  const float* wm2    = (const float*)d_in[8];
  const float* bm2    = (const float*)d_in[9];
  const float* pm2    = (const float*)d_in[10];
  const float* wa     = (const float*)d_in[11];
  const float* ba     = (const float*)d_in[12];
  const float* pa     = (const float*)d_in[13];
  const float* wf     = (const float*)d_in[14];
  const float* bf     = (const float*)d_in[15];
  float* out = (float*)d_out;

  const size_t ACT  = (size_t)NB * CC * HW;     // 2359296
  const size_t LEBW = (size_t)NRES * CC * KK;   // 589824
  const size_t QKE  = (size_t)NB * HW * 32;     // 589824

  char* ws = (char*)d_ws;
  size_t off = 0;
  auto alloc = [&](size_t bytes) -> char* {
    off = (off + 255) & ~(size_t)255;
    char* p = ws + off;
    off += bytes;
    return p;
  };
  _Float16* actA = (_Float16*)alloc(ACT * 2);
  _Float16* actB = (_Float16*)alloc(ACT * 2);
  _Float16* tmp  = (_Float16*)alloc(ACT * 2);
  _Float16* h2   = (_Float16*)alloc(ACT * 2);
  _Float16* qbuf = (_Float16*)alloc(QKE * 2);
  _Float16* kbuf = (_Float16*)alloc(QKE * 2);
  _Float16* vbuf = (_Float16*)alloc(ACT * 2);
  _Float16* wl1p = (_Float16*)alloc(LEBW * 2);
  _Float16* wl2p = (_Float16*)alloc(LEBW * 2);
  _Float16* wqp  = (_Float16*)alloc((size_t)CE * CC * 2);
  _Float16* wkp  = (_Float16*)alloc((size_t)CE * CC * 2);
  _Float16* wvp  = (_Float16*)alloc((size_t)CC * CC * 2);
  _Float16* wfp  = (_Float16*)alloc((size_t)CC * KK * 2);
  if (off > ws_size) return;  // workspace too small; bail deterministically

  auto cast = [&](const float* s, _Float16* d, int n) {
    cast_f32_to_f16<<<(n + 255) / 256, 256, 0, stream>>>(s, d, n);
  };
  auto repack = [&](const float* s, _Float16* d, int n) {
    repack_w3x3<<<(n + 255) / 256, 256, 0, stream>>>(s, d, n);
  };
  cast(x, actA, (int)ACT);
  repack(leb_w1, wl1p, (int)LEBW);     // 512 rows of (9*128)
  repack(leb_w2, wl2p, (int)LEBW);
  cast(wm1, wqp, CE * CC);
  cast(wm2, wkp, CE * CC);
  cast(wa, wvp, CC * CC);
  repack(wf, wfp, CC * KK);
  (void)hipMemsetAsync(qbuf, 0, QKE * 2, stream);  // zero ch-pad lanes 16..31
  (void)hipMemsetAsync(kbuf, 0, QKE * 2, stream);

  dim3 cgrid(NHW / 16, CC / 64);
  _Float16* ping = actA;
  _Float16* pong = actB;
  for (int i = 0; i < NRES; ++i) {
    conv3x3_wmma<<<cgrid, 128, 0, stream>>>(
        ping, wl1p + (size_t)i * CC * KK, leb_b1 + i * CC, MODE_RELU,
        nullptr, nullptr, tmp, nullptr);
    conv3x3_wmma<<<cgrid, 128, 0, stream>>>(
        tmp, wl2p + (size_t)i * CC * KK, leb_b2 + i * CC, MODE_RES,
        ping, nullptr, pong, nullptr);
    _Float16* sw = ping; ping = pong; pong = sw;
  }

  qkv_wmma<<<NHW / 16, 128, 0, stream>>>(ping, wqp, wkp, wvp, bm1, bm2, ba,
                                         pm1, pm2, pa, qbuf, kbuf, vbuf);

  const int attnLds = 16 * SRS * (int)sizeof(float);  // padded score strip
  (void)hipFuncSetAttribute((const void*)hspa_attn,
                            hipFuncAttributeMaxDynamicSharedMemorySize,
                            attnLds);
  hspa_attn<<<NB * (HW / 16), 128, attnLds, stream>>>(qbuf, kbuf, vbuf, ping, h2);

  conv3x3_wmma<<<cgrid, 128, 0, stream>>>(h2, wfp, bf, MODE_FINAL, nullptr, x,
                                          nullptr, out);
}
